// LSTMUnroll_24455543784002
// MI455X (gfx1250) — matrix operand order repaired and checked by reading, not compiled, not run on hardware
//
#include <hip/hip_runtime.h>
#include <hip/hip_bf16.h>

// ---- CDNA5 WMMA vector types (wave32) ----
typedef __attribute__((ext_vector_type(16))) __bf16 v16bf;
typedef __attribute__((ext_vector_type(8)))  __bf16 v8bf;
typedef __attribute__((ext_vector_type(8)))  float  v8f;
typedef __attribute__((ext_vector_type(4)))  float  v4f;

#define T_STEPS 30
#define BATCH   256
#define HID     1024
#define LAYERS  4
#define GATES   4
#define KTOT    (2 * HID)            // 2048 (x-kernel K then h-recurrent K)
#define NT_H    (HID / 16)           // 64 n-tiles per gate
#define KT_TOT  (KTOT / 32)          // 64 k-tiles of 32
#define BLOB    512                  // bf16 elements per 32x16 WMMA B fragment
#define PACK_PER_LAYER (GATES * NT_H * KT_TOT * BLOB)   // 8,388,608 elements
#define PACK_TOTAL     (LAYERS * PACK_PER_LAYER)        // 33,554,432 elements
#define BH (BATCH * HID)             // 262144

// -------------------------------------------------------------------------
// zero fp32 region (c-state + both h ping-pong buffers, contiguous in ws)
// -------------------------------------------------------------------------
__global__ void zero_f32(float* __restrict__ p, int n) {
    int i = blockIdx.x * blockDim.x + threadIdx.x;
    int stride = gridDim.x * blockDim.x;
    for (; i < n; i += stride) p[i] = 0.0f;
}

// -------------------------------------------------------------------------
// Pack fp32 [k ; rk] -> bf16 WMMA-B-fragment layout.
// Blob layout (512 bf16 = 1KB): element (lane, e):
//   offset = (e>=8 ? 256 : 0) + lane*8 + (e&7)      [bf16 units]
// B-matrix 32x16 (KxN), 16-bit: lanes 0-15: e0-7 -> K0-7, e8-15 -> K16-23,
// lanes 16-31: e0-7 -> K8-15, e8-15 -> K24-31; N = lane%16.
// Blob index: (((layer*4 + gate)*64 + nt)*64 + kt)
// -------------------------------------------------------------------------
__global__ void pack_weights(const float* __restrict__ kern,
                             const float* __restrict__ rkern,
                             __bf16* __restrict__ wp) {
    long o = (long)blockIdx.x * blockDim.x + threadIdx.x;   // < 33,554,432
    int idx  = (int)(o & 511);
    int kt   = (int)((o >> 9)  & 63);
    int nt   = (int)((o >> 15) & 63);
    int g    = (int)((o >> 21) & 3);
    int lyr  = (int)((o >> 23) & 3);
    int hib  = idx >> 8;                 // 0/1: e<8 or e>=8 half
    int lane = (idx & 255) >> 3;         // 0..31
    int e    = (idx & 7) + 8 * hib;      // 0..15
    int kk   = (e & 7) + 8 * (lane >> 4) + 16 * hib;     // K within 32-tile
    int n    = lane & 15;
    int K    = kt * 32 + kk;             // 0..2047
    int col  = g * HID + nt * 16 + n;    // 0..4095
    float v  = (K < HID)
        ? kern [((long)lyr * HID + K)         * (4 * HID) + col]
        : rkern[((long)lyr * HID + (K - HID)) * (4 * HID) + col];
    wp[o] = (__bf16)v;
}

// -------------------------------------------------------------------------
// Fused LSTM cell: z = [x;h] @ Wpack + b, gates, c/h update.
// Grid (8,8): blockIdx.x = 32-row M block, blockIdx.y = 128-col N block.
// 8 waves: wave w owns hidden cols [nblk*128 + w*16, +16), both 16-row
// sub-tiles of the 32-row strip, all 4 gate quadrants.
// A strip (32 x 2048) staged once in 128KB dynamic LDS as bf16 fragments.
// -------------------------------------------------------------------------
__device__ __forceinline__ float fsigmoid(float x) {
    return 1.0f / (1.0f + __expf(-x));
}
__device__ __forceinline__ float ftanh(float x) {
    return 1.0f - 2.0f / (__expf(2.0f * x) + 1.0f);
}

__global__ __launch_bounds__(256)
void lstm_cell(const float* __restrict__ xsrc,   // [256,1024] layer input
               const float* __restrict__ h_in,   // [256,1024] h(t-1)
               float*       __restrict__ h_out,  // [256,1024] h(t)
               float*       __restrict__ c_st,   // [256,1024] c in/out
               const __bf16* __restrict__ wp,    // this layer's packed W
               const float* __restrict__ bias)   // [4096]
{
    extern __shared__ __align__(16) char smem[];   // 128 KB: 64kt x 2mt x 1KB
    const int tid  = threadIdx.x;
    const int lane = tid & 31;
    const int wave = tid >> 5;
    const int mblk = blockIdx.x;    // 0..7
    const int nblk = blockIdx.y;    // 0..7

    // ---- stage A: rows [mblk*32, +32), K 0..2047 (x then h), bf16 frags ----
    // A-frag layout (16-bit 16x32): lane = (m%16) + 16*half,
    // half0: e0-7->K0-7, e8-15->K16-23 ; half1: e0-7->K8-15, e8-15->K24-31.
    // Blob bytes: (e>=8?512:0) + lane*16 + (e&7)*2  -> one 16B store per 8-K chunk.
    #pragma unroll 4
    for (int it = 0; it < 32; ++it) {
        const int ml = it;            // local row
        const int kg = tid * 8;       // 8-K chunk, coalesced across threads
        const float* src = (kg < HID)
            ? (xsrc + (size_t)(mblk * 32 + ml) * HID + kg)
            : (h_in + (size_t)(mblk * 32 + ml) * HID + (kg - HID));
        v4f f0 = *(const v4f*)src;
        v4f f1 = *(const v4f*)(src + 4);
        v8bf b;
        b[0] = (__bf16)f0[0]; b[1] = (__bf16)f0[1];
        b[2] = (__bf16)f0[2]; b[3] = (__bf16)f0[3];
        b[4] = (__bf16)f1[0]; b[5] = (__bf16)f1[1];
        b[6] = (__bf16)f1[2]; b[7] = (__bf16)f1[3];
        const int kt   = kg >> 5;
        const int kk0  = kg & 31;                // 0,8,16,24
        const int half = (kk0 >> 3) & 1;
        const int hib  = (kk0 & 16) ? 512 : 0;
        const int lns  = (ml & 15) + 16 * half;
        const int mt   = ml >> 4;
        *(v8bf*)(smem + ((((kt << 1) + mt) << 10) + hib + lns * 16)) = b;
    }
    __syncthreads();

    // ---- GEMM: 8 accumulators (4 gates x 2 m-tiles), K = 64 steps of 32 ----
    v8f acc[GATES][2] = {};
    const int    nt      = nblk * 8 + wave;                       // 0..63
    const size_t gstride = (size_t)NT_H * KT_TOT * BLOB;          // per-gate
    const __bf16* b0 = wp + (size_t)nt * KT_TOT * BLOB + lane * 8;

    #pragma unroll 2
    for (int kt = 0; kt < KT_TOT; ++kt) {
        // --- issue ALL loads for this K-step first (1 ds clause + 1 vmem
        //     clause of 8 b128s into distinct regs), WMMAs drain them with
        //     partial s_wait_loadcnt while earlier MMAs execute ---
        const char* ap = smem + ((kt << 1) << 10) + lane * 16;
        v8bf a0lo = *(const v8bf*)(ap);
        v8bf a0hi = *(const v8bf*)(ap + 512);
        v8bf a1lo = *(const v8bf*)(ap + 1024);
        v8bf a1hi = *(const v8bf*)(ap + 1536);

        v8bf blo[GATES], bhi[GATES];
        #pragma unroll
        for (int g = 0; g < GATES; ++g) {
            const __bf16* bp = b0 + (size_t)g * gstride + (size_t)kt * BLOB;
            blo[g] = *(const v8bf*)(bp);
            bhi[g] = *(const v8bf*)(bp + 256);
            // pull next tiles toward the WGP (near-resident hint, no guard:
            // speculative over-run past the last k-tiles is harmless)
            __builtin_prefetch(bp + 8 * BLOB, 0, 3);
        }

        v16bf A0 = __builtin_shufflevector(a0lo, a0hi,
                    0,1,2,3,4,5,6,7,8,9,10,11,12,13,14,15);
        v16bf A1 = __builtin_shufflevector(a1lo, a1hi,
                    0,1,2,3,4,5,6,7,8,9,10,11,12,13,14,15);
        #pragma unroll
        for (int g = 0; g < GATES; ++g) {
            v16bf B = __builtin_shufflevector(blo[g], bhi[g],
                       0,1,2,3,4,5,6,7,8,9,10,11,12,13,14,15);
            acc[g][0] = __builtin_amdgcn_wmma_f32_16x16x32_bf16(
                false, A0, false, B, (short)0, acc[g][0], false, false);
            acc[g][1] = __builtin_amdgcn_wmma_f32_16x16x32_bf16(
                false, A1, false, B, (short)0, acc[g][1], false, false);
        }
    }

    // ---- epilogue: gates + state update, fully fused, fp32 ----
    // C/D layout: lane l -> N = l%16, rows M = 8*(l/16) + r (r = VGPR 0..7).
    const int nl   = lane & 15;
    const int ncol = nblk * 128 + wave * 16 + nl;    // 0..1023
    float bz[GATES];
    #pragma unroll
    for (int g = 0; g < GATES; ++g) bz[g] = bias[g * HID + ncol];

    #pragma unroll
    for (int mt = 0; mt < 2; ++mt) {
        const int mbase = mblk * 32 + mt * 16 + ((lane >> 4) << 3);
        #pragma unroll
        for (int r = 0; r < 8; ++r) {
            const size_t off = (size_t)(mbase + r) * HID + ncol;
            float zi = acc[0][mt][r] + bz[0];
            float zf = acc[1][mt][r] + bz[1];
            float zg = acc[2][mt][r] + bz[2];
            float zo = acc[3][mt][r] + bz[3];
            float cold = c_st[off];
            float cn = fsigmoid(zf) * cold + fsigmoid(zi) * ftanh(zg);
            float hn = fsigmoid(zo) * ftanh(cn);
            c_st[off]  = cn;     // also feeds next layer as "cur"
            h_out[off] = hn;
        }
    }
}

// -------------------------------------------------------------------------
extern "C" void kernel_launch(void* const* d_in, const int* in_sizes, int n_in,
                              void* d_out, int out_size, void* d_ws, size_t ws_size,
                              hipStream_t stream) {
    const float* inputs = (const float*)d_in[0];   // [30,256,1024]
    const float* kern   = (const float*)d_in[1];   // [4,1024,4096]
    const float* rkern  = (const float*)d_in[2];   // [4,1024,4096]
    const float* bias   = (const float*)d_in[3];   // [4,4096]

    char* ws = (char*)d_ws;
    __bf16* wpack = (__bf16*)ws;                                   // 64 MB
    float*  c_st  = (float*)(ws + (size_t)PACK_TOTAL * 2);         // 4 MB
    float*  h_a   = c_st + (size_t)LAYERS * BH;                    // 4 MB
    float*  h_b   = h_a  + (size_t)LAYERS * BH;                    // 4 MB

    // zero c + both h buffers (contiguous): 3 * 4 * 256 * 1024 floats
    zero_f32<<<2048, 256, 0, stream>>>(c_st, 3 * LAYERS * BH);

    // pack weights to bf16 WMMA fragments (L2-resident thereafter)
    pack_weights<<<PACK_TOTAL / 256, 256, 0, stream>>>(kern, rkern, wpack);

    const float* h_read = h_a;
    float*       h_write = h_b;
    for (int t = 0; t < T_STEPS; ++t) {
        for (int j = 0; j < LAYERS; ++j) {
            const float* xsrc = (j == 0)
                ? (inputs + (size_t)t * BH)
                : (c_st + (size_t)(j - 1) * BH);   // cur = c_new of layer j-1
            lstm_cell<<<dim3(8, 8), 256, 128 * 1024, stream>>>(
                xsrc,
                h_read  + (size_t)j * BH,
                h_write + (size_t)j * BH,
                c_st    + (size_t)j * BH,
                wpack   + (size_t)j * PACK_PER_LAYER,
                bias    + (size_t)j * (4 * HID));
        }
        const float* tmp = h_read;
        h_read  = h_write;
        h_write = (float*)tmp;
    }

    // final output: c-state of last layer after last timestep
    hipMemcpyAsync(d_out, c_st + (size_t)(LAYERS - 1) * BH,
                   (size_t)BH * sizeof(float),
                   hipMemcpyDeviceToDevice, stream);
}
